// GroupedQueryAttention_31129922961571
// MI455X (gfx1250) — matrix operand order repaired and checked by reading, not compiled, not run on hardware
//
#include <hip/hip_runtime.h>

// ---------------------------------------------------------------------------
// MI455X / gfx1250 GQA forward — wave32, v_wmma_f32_16x16x32_bf16, f32 accum.
// Pipeline:
//   1) Qb  = bf16(x @ wq + bq)             [4096,2048]       (GEMM mode 0)
//   2) Kb  = bf16(x @ wk + bk)             [4096, 512]       (GEMM mode 0)
//   3) Vt  = bf16(x @ wv + bv)^T per group [4,8,64,1024]     (GEMM mode 1)
//   4) attn = softmax(Q K^T / 8) -> d_out attn region (fp32, required output)
//   5) O   = attn @ V (fp32)               [4096,2048]
//   6) out = O @ fc + bfc -> d_out         (GEMM mode 2)
// GEMM: 64x128x32 block tile, double-buffered B in LDS, A fragments loaded
// straight from global (32B vector loads + in-register bf16 cvt).
// ---------------------------------------------------------------------------

typedef __bf16 bf16;
typedef __attribute__((ext_vector_type(16))) __bf16 v16bf;
typedef __attribute__((ext_vector_type(8)))  __bf16 bf16x8;
typedef __attribute__((ext_vector_type(8)))  float  v8f;
typedef __attribute__((ext_vector_type(8)))  float  f32x8;
typedef __attribute__((ext_vector_type(4)))  float  f32x4;

#define D_MODEL   2048
#define S_LEN     1024
#define BATCH     4
#define HEADS     32
#define GROUPS    8
#define HEAD_DIM  64
#define GROUP_DIM 512
#define M_TOTAL   4096   // B*S

#define BM 64
#define BN 128
#define BK 32

__device__ __forceinline__ bf16x8 cvt8(f32x8 v) {
    bf16x8 r;
#pragma unroll
    for (int i = 0; i < 8; ++i) r[i] = (bf16)v[i];
    return r;
}

// ---------------------------------------------------------------------------
// C[M,N] = A[M,K](fp32) * W[K,N](fp32) + bias, bf16 WMMA core.
// 256 threads = 8 waves; wave (wr = w&3, wc = w>>2) owns 16 rows x 64 cols
// (4 accumulators). mode 0: bf16 row out; mode 1: bf16 V-transposed out;
// mode 2: fp32 row out.
// ---------------------------------------------------------------------------
__global__ __launch_bounds__(256)
void gemm_bias_kernel(const float* __restrict__ A, int lda,
                      const float* __restrict__ W, int ldw,
                      const float* __restrict__ bias,
                      int N, int K,
                      void* __restrict__ out, int mode)
{
    __shared__ __align__(16) bf16 Bs[2][BN][BK];   // transposed: [n][k]

    const int tid  = threadIdx.x;
    const int lane = tid & 31;
    const int w    = tid >> 5;
    const int half = lane >> 4;
    const int l15  = lane & 15;
    const int bm   = blockIdx.x * BM;
    const int bn   = blockIdx.y * BN;
    const int wr   = w & 3;     // 16-row tile
    const int wc   = w >> 2;    // 64-col strip

    // B staging: thread handles k-row skk, 16 n's starting at snb
    const int skk = tid >> 3;          // 0..31
    const int snb = (tid & 7) * 16;    // 0..112

    v8f acc[4] = {};

    const int NT = K / BK;

    // ---- stage tile 0 ----
    {
        const float* src = W + (size_t)skk * ldw + bn + snb;
        bf16x8 b0 = cvt8(*(const f32x8*)src);
        bf16x8 b1 = cvt8(*(const f32x8*)(src + 8));
#pragma unroll
        for (int i = 0; i < 8; ++i) {
            Bs[0][snb + i][skk]     = b0[i];
            Bs[0][snb + 8 + i][skk] = b1[i];
        }
    }
    __syncthreads();

    const float* arow = A + (size_t)(bm + wr * 16 + l15) * lda;

#pragma unroll 1
    for (int kt = 0; kt < NT; ++kt) {
        const int cur = kt & 1;

        // ---- prefetch + stage next B tile into the other buffer ----
        if (kt + 1 < NT) {
            const float* src = W + (size_t)((kt + 1) * BK + skk) * ldw + bn + snb;
            __builtin_prefetch(src, 0, 1);                 // global_prefetch_b8
            bf16x8 b0 = cvt8(*(const f32x8*)src);
            bf16x8 b1 = cvt8(*(const f32x8*)(src + 8));
#pragma unroll
            for (int i = 0; i < 8; ++i) {
                Bs[cur ^ 1][snb + i][skk]     = b0[i];
                Bs[cur ^ 1][snb + 8 + i][skk] = b1[i];
            }
        }

        // ---- A fragment straight from global: K = half*16 .. +15 (contiguous) ----
        v16bf af;
        {
            const float* p = arow + kt * BK + half * 16;
            ((bf16x8*)&af)[0] = cvt8(*(const f32x8*)p);
            ((bf16x8*)&af)[1] = cvt8(*(const f32x8*)(p + 8));
        }

        // ---- 4 B fragments + 4 WMMAs ----
#pragma unroll
        for (int t = 0; t < 4; ++t) {
            v16bf bfrag;  // lane n = l15, K = half*16 .. +15 (contiguous)
            const bf16* p = &Bs[cur][wc * 64 + t * 16 + l15][half * 16];
            ((bf16x8*)&bfrag)[0] = *(const bf16x8*)p;
            ((bf16x8*)&bfrag)[1] = *(const bf16x8*)(p + 8);
            acc[t] = __builtin_amdgcn_wmma_f32_16x16x32_bf16(
                false, af, false, bfrag, (short)0, acc[t], false, false);
        }
        __syncthreads();
    }

    // ---- epilogue: C/D layout m = half*8 + r, n = l15 ----
    const int rbase = bm + wr * 16 + half * 8;
#pragma unroll
    for (int t = 0; t < 4; ++t) {
        const int col  = bn + wc * 64 + t * 16 + l15;
        const float bv = bias[col];
        if (mode == 0) {
            bf16* o = (bf16*)out;
#pragma unroll
            for (int r = 0; r < 8; ++r)
                o[(size_t)(rbase + r) * N + col] = (bf16)(acc[t][r] + bv);
        } else if (mode == 1) {
            // Vt[((b*8+g)*64+hd)*1024 + s]; rows rbase..rbase+7 -> contiguous s
            bf16* o = (bf16*)out;
            const int g = col >> 6, hd = col & 63;
            const int b = rbase >> 10, s = rbase & 1023;
            bf16x8 pk;
#pragma unroll
            for (int r = 0; r < 8; ++r) pk[r] = (bf16)(acc[t][r] + bv);
            *(bf16x8*)&o[(size_t)((b * 8 + g) * 64 + hd) * 1024 + s] = pk;
        } else {
            float* o = (float*)out;
#pragma unroll
            for (int r = 0; r < 8; ++r)
                o[(size_t)(rbase + r) * N + col] = acc[t][r] + bv;
        }
    }
}

// ---------------------------------------------------------------------------
// scores + softmax: one block per (q16-tile, head, batch). 8 waves; wave w
// computes key-tiles w, w+8, ..., w+56 (hd=64 -> two 32-wide k-steps).
// 16x1024 fp32 score strip in LDS (64 KB of the 320 KB WGP pool), then
// per-row softmax with __shfl_xor reductions and float4 attn writes.
// ---------------------------------------------------------------------------
__global__ __launch_bounds__(256)
void scores_softmax_kernel(const bf16* __restrict__ Qb,   // [4096,2048]
                           const bf16* __restrict__ Kb,   // [4096, 512]
                           float* __restrict__ attn)      // [4,32,1024,1024]
{
    __shared__ __align__(16) float sc[16][1024];

    const int tid  = threadIdx.x;
    const int lane = tid & 31;
    const int w    = tid >> 5;
    const int half = lane >> 4;
    const int l15  = lane & 15;
    const int qt = blockIdx.x;            // 0..63
    const int h  = blockIdx.y;            // 0..31
    const int b  = blockIdx.z;            // 0..3
    const int g  = h >> 2;                // GQA: head h uses group h/REP
    const int q0 = qt * 16;

    // Q fragments for this q-tile (reused by all 8 key tiles of this wave)
    v16bf aq[2];
    {
        const bf16* qrow = Qb + (size_t)(b * S_LEN + q0 + l15) * D_MODEL + h * HEAD_DIM;
#pragma unroll
        for (int ks = 0; ks < 2; ++ks) {
            const bf16* p = qrow + ks * 32 + half * 16;
            ((bf16x8*)&aq[ks])[0] = *(const bf16x8*)p;
            ((bf16x8*)&aq[ks])[1] = *(const bf16x8*)(p + 8);
        }
    }

#pragma unroll 1
    for (int kt = w; kt < 64; kt += 8) {
        v8f acc = {};
#pragma unroll
        for (int ks = 0; ks < 2; ++ks) {
            // B = K^T: lane n = key (kt*16+l15), 16 contiguous hd values (32B)
            const bf16* p = Kb + (size_t)(b * S_LEN + kt * 16 + l15) * GROUP_DIM
                            + g * HEAD_DIM + ks * 32 + half * 16;
            v16bf bk = *(const v16bf*)p;
            acc = __builtin_amdgcn_wmma_f32_16x16x32_bf16(
                false, aq[ks], false, bk, (short)0, acc, false, false);
        }
#pragma unroll
        for (int r = 0; r < 8; ++r)
            sc[half * 8 + r][kt * 16 + l15] = acc[r] * 0.125f;  // 1/sqrt(64)
    }
    __syncthreads();

    // softmax: wave w handles rows 2w and 2w+1
    float* attn_base = attn + ((size_t)(b * HEADS + h) * S_LEN + q0) * S_LEN;
#pragma unroll
    for (int rr = 0; rr < 2; ++rr) {
        const int row = w * 2 + rr;
        float mx = -3.0e38f;
        for (int k = lane; k < 1024; k += 32) mx = fmaxf(mx, sc[row][k]);
#pragma unroll
        for (int off = 16; off; off >>= 1) mx = fmaxf(mx, __shfl_xor(mx, off, 32));
        float sum = 0.f;
        for (int k = lane; k < 1024; k += 32) {
            const float e = __expf(sc[row][k] - mx);
            sc[row][k] = e;
            sum += e;
        }
#pragma unroll
        for (int off = 16; off; off >>= 1) sum += __shfl_xor(sum, off, 32);
        const float inv = 1.f / sum;
        float* orow = attn_base + (size_t)row * S_LEN;
        for (int k4 = lane; k4 < 256; k4 += 32) {       // float4 stores
            f32x4 v = *(const f32x4*)&sc[row][k4 * 4];
            v *= inv;
            *(f32x4*)&orow[k4 * 4] = v;
        }
    }
}

// ---------------------------------------------------------------------------
// O = attn @ V. Block per (32-row q-tile, head, batch); 8 waves cover
// 32(M) x 64(N=hd). A = attn fp32 (f32x8 loads, cvt->bf16), B = V^T bf16
// (one 32B contiguous load per fragment).
// ---------------------------------------------------------------------------
__global__ __launch_bounds__(256)
void av_kernel(const float* __restrict__ attn,   // [4,32,1024,1024]
               const bf16* __restrict__ Vt,      // [4,8,64,1024]
               float* __restrict__ Ob)           // [4096,2048] fp32
{
    const int tid  = threadIdx.x;
    const int lane = tid & 31;
    const int w    = tid >> 5;
    const int half = lane >> 4;
    const int l15  = lane & 15;
    const int qt = blockIdx.x;            // 0..31
    const int h  = blockIdx.y;
    const int b  = blockIdx.z;
    const int g  = h >> 2;
    const int wr = w & 1;                 // 2 row tiles of 16
    const int wc = w >> 1;                // 4 col tiles of 16 (hd)
    const int q0 = qt * 32 + wr * 16;

    v8f acc = {};
    const float* arow = attn + ((size_t)(b * HEADS + h) * S_LEN + q0 + l15) * S_LEN;
    const bf16*  vrow = Vt + ((size_t)(b * GROUPS + g) * HEAD_DIM + wc * 16 + l15) * S_LEN;

#pragma unroll 1
    for (int kk = 0; kk < S_LEN; kk += 32) {
        v16bf af;   // K = half*16 .. +15 (contiguous)
        ((bf16x8*)&af)[0] = cvt8(*(const f32x8*)(arow + kk + half * 16));
        ((bf16x8*)&af)[1] = cvt8(*(const f32x8*)(arow + kk + half * 16 + 8));

        v16bf bv = *(const v16bf*)(vrow + kk + half * 16);  // 32B contiguous
        acc = __builtin_amdgcn_wmma_f32_16x16x32_bf16(
            false, af, false, bv, (short)0, acc, false, false);
    }

    const int col   = h * HEAD_DIM + wc * 16 + l15;
    const int rbase = b * S_LEN + qt * 32 + wr * 16 + half * 8;
#pragma unroll
    for (int r = 0; r < 8; ++r)
        Ob[(size_t)(rbase + r) * D_MODEL + col] = acc[r];
}

// ---------------------------------------------------------------------------
extern "C" void kernel_launch(void* const* d_in, const int* in_sizes, int n_in,
                              void* d_out, int out_size, void* d_ws, size_t ws_size,
                              hipStream_t stream)
{
    (void)in_sizes; (void)n_in; (void)out_size; (void)ws_size;

    const float* x    = (const float*)d_in[0];
    const float* wq_w = (const float*)d_in[1];
    const float* wq_b = (const float*)d_in[2];
    const float* wk_w = (const float*)d_in[3];
    const float* wk_b = (const float*)d_in[4];
    const float* wv_w = (const float*)d_in[5];
    const float* wv_b = (const float*)d_in[6];
    const float* fc_w = (const float*)d_in[7];
    const float* fc_b = (const float*)d_in[8];

    float* out  = (float*)d_out;                               // [4,1024,2048]
    float* attn = out + (size_t)BATCH * S_LEN * D_MODEL;       // [4,32,1024,1024]

    char* ws = (char*)d_ws;                  // 56 MB total
    bf16*  Qb = (bf16*)(ws);                 // 16 MB  [4096,2048]
    bf16*  Kb = (bf16*)(ws + (16u << 20));   //  4 MB  [4096,512]
    bf16*  Vt = (bf16*)(ws + (20u << 20));   //  4 MB  [4,8,64,1024]
    float* Ob = (float*)(ws + (24u << 20));  // 32 MB  [4096,2048]

    const dim3 blk(256);

    // projections
    gemm_bias_kernel<<<dim3(M_TOTAL / BM, D_MODEL / BN), blk, 0, stream>>>(
        x, D_MODEL, wq_w, D_MODEL, wq_b, D_MODEL, D_MODEL, Qb, 0);
    gemm_bias_kernel<<<dim3(M_TOTAL / BM, GROUP_DIM / BN), blk, 0, stream>>>(
        x, D_MODEL, wk_w, GROUP_DIM, wk_b, GROUP_DIM, D_MODEL, Kb, 0);
    gemm_bias_kernel<<<dim3(M_TOTAL / BM, GROUP_DIM / BN), blk, 0, stream>>>(
        x, D_MODEL, wv_w, GROUP_DIM, wv_b, GROUP_DIM, D_MODEL, Vt, 1);

    // attention
    scores_softmax_kernel<<<dim3(S_LEN / 16, HEADS, BATCH), blk, 0, stream>>>(Qb, Kb, attn);
    av_kernel<<<dim3(S_LEN / 32, HEADS, BATCH), blk, 0, stream>>>(attn, Vt, Ob);

    // output projection
    gemm_bias_kernel<<<dim3(M_TOTAL / BM, D_MODEL / BN), blk, 0, stream>>>(
        Ob, D_MODEL, fc_w, D_MODEL, fc_b, D_MODEL, D_MODEL, out, 2);
}
